// MambaBlock_1649267442293
// MI455X (gfx1250) — matrix operand order repaired and checked by reading, not compiled, not run on hardware
//
#include <hip/hip_runtime.h>
#include <cstdint>
#include <cstddef>

// ---------------------------------------------------------------------------
// Bidirectional Mamba block for MI455X (gfx1250), wave32, bf16 WMMA GEMMs.
// GEMMs: 32x64 macro-tile per wave (8 v_wmma / K-step, 1.5 b128 loads/wmma),
// with compile-time LDA/K/LDC so all fragment addresses fold into the 24-bit
// instruction offsets (2 live base pointers, no scratch spills).
// ---------------------------------------------------------------------------

typedef __attribute__((ext_vector_type(16))) __bf16        bf16x16;
typedef __attribute__((ext_vector_type(8)))  float         f32x8;
typedef __attribute__((ext_vector_type(4)))  unsigned int  u32x4;

#define DEVFN __device__ __forceinline__

constexpr int BB   = 4;
constexpr int LL   = 2048;
constexpr int DM   = 512;            // d_model
constexpr int DI   = 1024;           // d_inner
constexpr int DS   = 16;             // d_state
constexpr int DTR  = 32;             // dt_rank
constexpr int XPN  = DTR + 2 * DS;   // 64 x_proj output cols
constexpr int MTOK = BB * LL;        // 8192 tokens

// ---- bf16 <-> f32 helpers (storage = unsigned short) ----------------------
DEVFN float bf2f(unsigned short u) {
    unsigned int x = ((unsigned int)u) << 16;
    float f; __builtin_memcpy(&f, &x, 4); return f;
}
DEVFN unsigned short f2bf(float f) {
    unsigned int x; __builtin_memcpy(&x, &f, 4);
    x += 0x7FFFu + ((x >> 16) & 1u);           // round-to-nearest-even
    return (unsigned short)(x >> 16);
}
DEVFN float siluf(float x) { return x * (1.0f / (1.0f + __expf(-x))); }
DEVFN float softplusf(float x) { return x > 20.0f ? x : __logf(1.0f + __expf(x)); }

// ---------------------------------------------------------------------------
// Kernel: f32 -> bf16 conversion (weights / packing)
// ---------------------------------------------------------------------------
__global__ void cvt_f32_bf16(const float* __restrict__ s, unsigned short* __restrict__ d, int n) {
    int i = blockIdx.x * 256 + threadIdx.x;
    if (i < n) d[i] = f2bf(s[i]);
}

// ---------------------------------------------------------------------------
// Kernel: LayerNorm (one wave per token, 512 = 16 elems/lane) + residual copy
// ---------------------------------------------------------------------------
__global__ void layernorm_kernel(const float* __restrict__ hs,
                                 const float* __restrict__ nw,
                                 const float* __restrict__ nb,
                                 unsigned short* __restrict__ hn,
                                 float* __restrict__ resid) {
    int wave = threadIdx.x >> 5;
    int lane = threadIdx.x & 31;
    int tok  = blockIdx.x * 8 + wave;
    const float* row = hs + (size_t)tok * DM;

    float vals[16];
    float s = 0.f, ss = 0.f;
#pragma unroll
    for (int i = 0; i < 16; ++i) {
        float v = row[lane + i * 32];
        vals[i] = v; s += v; ss += v * v;
    }
#pragma unroll
    for (int m = 16; m >= 1; m >>= 1) {
        s  += __shfl_xor(s,  m, 32);
        ss += __shfl_xor(ss, m, 32);
    }
    float mu   = s * (1.0f / DM);
    float var  = ss * (1.0f / DM) - mu * mu;
    float rstd = rsqrtf(var + 1e-5f);

    size_t base = (size_t)tok * DM;
#pragma unroll
    for (int i = 0; i < 16; ++i) {
        int c = lane + i * 32;
        float hv = (vals[i] - mu) * rstd * nw[c] + nb[c];
        hn[base + c]    = f2bf(hv);
        resid[base + c] = vals[i];                 // second output of the tuple
    }
}

// ---------------------------------------------------------------------------
// Kernel: bf16 WMMA GEMM.  C[M,N] = scale * A[M,K] * W[N,K]^T
// One wave computes a (16*TM) x (16*TN) macro-tile: TM*TN accumulators
// sharing TM A-fragments and TN B-fragments per 32-deep K step.
// LDA/K/LDC are compile-time so all sub-tile strides fold into the 24-bit
// global_load instruction offsets: only 2 base pointers stay live.
// Operand layouts follow CDNA5 ISA 7.12.2 (16-bit A 16x32 / B 32x16).
// 8 waves per block; grid sized exactly so EXEC is all-ones for every WMMA.
// ---------------------------------------------------------------------------
template <int TM, int TN, int LDA, int K, int LDC, bool OUT_BF16>
__global__ __launch_bounds__(256)
void gemm_wmma_bf16(const unsigned short* __restrict__ A,
                    const unsigned short* __restrict__ W,   // [N,K] row-major
                    void* __restrict__ Cp,
                    int ntn,                                 // N / (16*TN)
                    float scale) {
    int wave = threadIdx.x >> 5;
    int lane = threadIdx.x & 31;
    int tile = blockIdx.x * 8 + wave;
    int tm   = tile / ntn;
    int tn   = tile - tm * ntn;

    int m0   = tm * 16 * TM;
    int n0   = tn * 16 * TN;
    int half = lane >> 4;                          // 0: lanes 0-15, 1: lanes 16-31
    int l16  = lane & 15;

    // Single base pointer per operand; all other offsets are compile-time.
    const unsigned short* Ab = A + (size_t)(m0 + l16) * LDA + half * 8;
    const unsigned short* Wb = W + (size_t)(n0 + l16) * K   + half * 8;

    f32x8 acc[TM][TN];
#pragma unroll
    for (int i = 0; i < TM; ++i)
#pragma unroll
        for (int j = 0; j < TN; ++j)
            acc[i][j] = (f32x8){0.f, 0.f, 0.f, 0.f, 0.f, 0.f, 0.f, 0.f};

    union Frag { bf16x16 v; u32x4 q[2]; };

#pragma unroll 2
    for (int k = 0; k < K; k += 32) {
        Frag a[TM], b[TN];
        // A 16x32 bf16: lanes0-15 hold K k..k+7 & k+16..k+23 ; lanes16-31 +8
#pragma unroll
        for (int i = 0; i < TM; ++i) {
            a[i].q[0] = *(const u32x4*)(Ab + i * 16 * LDA + k);
            a[i].q[1] = *(const u32x4*)(Ab + i * 16 * LDA + k + 16);
        }
        // B 32x16 bf16: lane = N column; lanes0-15 K k..k+7 & k+16..k+23 ; lanes16-31 +8
#pragma unroll
        for (int j = 0; j < TN; ++j) {
            b[j].q[0] = *(const u32x4*)(Wb + j * 16 * K + k);
            b[j].q[1] = *(const u32x4*)(Wb + j * 16 * K + k + 16);
        }
        if (K > 64) {                                  // speculative; OOB dropped
            __builtin_prefetch(Ab + k + 64, 0, 0);     // global_prefetch
            __builtin_prefetch(Wb + k + 64, 0, 0);
        }
#pragma unroll
        for (int i = 0; i < TM; ++i)
#pragma unroll
            for (int j = 0; j < TN; ++j)
                acc[i][j] = __builtin_amdgcn_wmma_f32_16x16x32_bf16(
                    false, a[i].v, false, b[j].v, (short)0, acc[i][j], false, false);
    }

    // C 16x16 f32: VGPR r -> (row +r, lanes0-15) / (row +8+r, lanes16-31)
#pragma unroll
    for (int i = 0; i < TM; ++i)
#pragma unroll
        for (int j = 0; j < TN; ++j)
#pragma unroll
            for (int r = 0; r < 8; ++r) {
                int row = m0 + i * 16 + r + half * 8;
                int col = n0 + j * 16 + l16;
                float v = acc[i][j][r] * scale;
                if (OUT_BF16) ((unsigned short*)Cp)[(size_t)row * LDC + col] = f2bf(v);
                else          ((float*)Cp)[(size_t)row * LDC + col] = v;
            }
}

// ---------------------------------------------------------------------------
// Kernel: causal depthwise conv (k=4) + bias + SiLU, with optional seq reverse
// xz bf16 [MTOK, 2*DI] (x = cols 0..DI-1) -> xc bf16 [MTOK, DI] in LOGICAL order
// ---------------------------------------------------------------------------
__global__ void conv_silu_kernel(const unsigned short* __restrict__ xz,
                                 const float* __restrict__ w,   // [DI,4]
                                 const float* __restrict__ bias,
                                 unsigned short* __restrict__ xc,
                                 int reverse) {
    size_t i = (size_t)blockIdx.x * 256 + threadIdx.x;   // over MTOK*DI
    int    d  = (int)(i & (DI - 1));
    size_t bt = i >> 10;
    int    t  = (int)(bt & (LL - 1));
    int    b  = (int)(bt >> 11);

    float acc = bias[d];
#pragma unroll
    for (int j = 0; j < 4; ++j) {
        int u = t - 3 + j;                                // logical position
        if (u >= 0) {
            int phys = reverse ? (LL - 1 - u) : u;
            acc += w[d * 4 + j] *
                   bf2f(xz[((size_t)(b * LL + phys)) * (2 * DI) + d]);
        }
    }
    xc[bt * DI + d] = f2bf(siluf(acc));
}

// ---------------------------------------------------------------------------
// Kernel: pack x_dbl[:, :DTR] (f32) -> bf16 A-matrix for the dt GEMM
// ---------------------------------------------------------------------------
__global__ void pack_dtin_kernel(const float* __restrict__ xdbl,
                                 unsigned short* __restrict__ dtin) {
    size_t i   = (size_t)blockIdx.x * 256 + threadIdx.x;  // MTOK*DTR
    size_t tok = i >> 5;
    int    r   = (int)(i & 31);
    dtin[i] = f2bf(xdbl[tok * XPN + r]);
}

// ---------------------------------------------------------------------------
// Kernel: selective scan. 4 lanes per (b,d), each owning 4 states.
// dt_ys: bf16 raw dt in, y (scan + D*skip) out IN-PLACE at logical index.
// ---------------------------------------------------------------------------
__global__ void scan_kernel(const unsigned short* __restrict__ xc,   // [MTOK,DI] bf16
                            unsigned short* __restrict__ dt_ys,      // [MTOK,DI] bf16
                            const float* __restrict__ xdbl,          // [MTOK,64] f32
                            const float* __restrict__ dt_bias,       // [DI]
                            const float* __restrict__ A_log,         // [DI,16]
                            const float* __restrict__ D_skip) {      // [DI]
    int tid = blockIdx.x * 256 + threadIdx.x;   // 16384 threads
    int sg  = tid & 3;
    int d   = (tid >> 2) & (DI - 1);
    int b   = tid >> 12;

    float Aa[4];
#pragma unroll
    for (int s = 0; s < 4; ++s) Aa[s] = -__expf(A_log[d * DS + sg * 4 + s]);
    float bias = dt_bias[d];
    float Dk   = D_skip[d];
    float h0 = 0.f, h1 = 0.f, h2 = 0.f, h3 = 0.f;

    size_t base = (size_t)b * LL * DI + d;
    size_t bc   = (size_t)b * LL * XPN;

    for (int t = 0; t < LL; ++t) {
        size_t idx = base + (size_t)t * DI;
        float xv  = bf2f(xc[idx]);
        float dtv = softplusf(bf2f(dt_ys[idx]) + bias);
        float xdt = dtv * xv;
        const float4 Bv = *(const float4*)(xdbl + bc + (size_t)t * XPN + DTR      + sg * 4);
        const float4 Cv = *(const float4*)(xdbl + bc + (size_t)t * XPN + DTR + DS + sg * 4);

        h0 = __expf(dtv * Aa[0]) * h0 + xdt * Bv.x; float y = h0 * Cv.x;
        h1 = __expf(dtv * Aa[1]) * h1 + xdt * Bv.y; y += h1 * Cv.y;
        h2 = __expf(dtv * Aa[2]) * h2 + xdt * Bv.z; y += h2 * Cv.z;
        h3 = __expf(dtv * Aa[3]) * h3 + xdt * Bv.w; y += h3 * Cv.w;

        y += __shfl_xor(y, 1, 32);                  // reduce across the 4-lane
        y += __shfl_xor(y, 2, 32);                  // state group
        if (sg == 0) dt_ys[idx] = f2bf(y + xv * Dk);
    }
}

// ---------------------------------------------------------------------------
// Kernel: combine  ysum = (ys_f[tok] + ys_b[rev(tok)]) * silu(z[tok])  (bf16)
// ---------------------------------------------------------------------------
__global__ void combine_kernel(const unsigned short* __restrict__ ysf,
                               const unsigned short* __restrict__ ysb,
                               const unsigned short* __restrict__ xz,
                               unsigned short* __restrict__ ysum) {
    size_t i  = (size_t)blockIdx.x * 256 + threadIdx.x;  // MTOK*DI
    int    d  = (int)(i & (DI - 1));
    size_t bt = i >> 10;
    int    t  = (int)(bt & (LL - 1));
    size_t b  = bt >> 11;
    size_t rt = b * LL + (LL - 1 - t);

    float z = bf2f(xz[bt * (2 * DI) + DI + d]);
    float v = (bf2f(ysf[i]) + bf2f(ysb[rt * DI + d])) * siluf(z);
    ysum[i] = f2bf(v);
}

// ---------------------------------------------------------------------------
// Host side
// ---------------------------------------------------------------------------
extern "C" void kernel_launch(void* const* d_in, const int* in_sizes, int n_in,
                              void* d_out, int out_size, void* d_ws, size_t ws_size,
                              hipStream_t stream) {
    const float* hs    = (const float*)d_in[0];
    const float* nw    = (const float*)d_in[1];
    const float* nb    = (const float*)d_in[2];
    const float* w_in  = (const float*)d_in[3];   // [2*DI, DM]
    const float* w_out = (const float*)d_in[4];   // [DM, DI]
    const float* cw_f  = (const float*)d_in[5];
    const float* cb_f  = (const float*)d_in[6];
    const float* xp_f  = (const float*)d_in[7];   // [64, DI]
    const float* dtw_f = (const float*)d_in[8];   // [DI, 32]
    const float* dtb_f = (const float*)d_in[9];
    const float* Al_f  = (const float*)d_in[10];
    const float* Dk_f  = (const float*)d_in[11];
    const float* cw_b  = (const float*)d_in[12];
    const float* cb_b  = (const float*)d_in[13];
    const float* xp_b  = (const float*)d_in[14];
    const float* dtw_b = (const float*)d_in[15];
    const float* dtb_b = (const float*)d_in[16];
    const float* Al_b  = (const float*)d_in[17];
    const float* Dk_b  = (const float*)d_in[18];

    float* out_p   = (float*)d_out;                       // [MTOK, DM]
    float* resid_p = (float*)d_out + (size_t)MTOK * DM;   // [MTOK, DM]

    // ---- scratch carve-out ----
    char* p = (char*)d_ws;
    auto alloc = [&](size_t bytes) -> char* {
        char* r = p; p += (bytes + 255) & ~(size_t)255; return r;
    };
    unsigned short* HN    = (unsigned short*)alloc((size_t)MTOK * DM  * 2);  // ln out bf16
    unsigned short* WIN   = (unsigned short*)alloc((size_t)2*DI * DM  * 2);
    unsigned short* WOUT  = (unsigned short*)alloc((size_t)DM   * DI  * 2);
    unsigned short* WXPF  = (unsigned short*)alloc((size_t)XPN  * DI  * 2);
    unsigned short* WXPB  = (unsigned short*)alloc((size_t)XPN  * DI  * 2);
    unsigned short* WDTF  = (unsigned short*)alloc((size_t)DI   * DTR * 2);
    unsigned short* WDTB  = (unsigned short*)alloc((size_t)DI   * DTR * 2);
    unsigned short* XZ    = (unsigned short*)alloc((size_t)MTOK * 2*DI* 2);
    unsigned short* XCF   = (unsigned short*)alloc((size_t)MTOK * DI  * 2);
    unsigned short* XCB   = (unsigned short*)alloc((size_t)MTOK * DI  * 2);
    float*          XDBLF = (float*)         alloc((size_t)MTOK * XPN * 4);
    float*          XDBLB = (float*)         alloc((size_t)MTOK * XPN * 4);
    unsigned short* DTINF = (unsigned short*)alloc((size_t)MTOK * DTR * 2);
    unsigned short* DTINB = (unsigned short*)alloc((size_t)MTOK * DTR * 2);
    unsigned short* DTF   = (unsigned short*)alloc((size_t)MTOK * DI  * 2);  // dt -> ys
    unsigned short* DTB   = (unsigned short*)alloc((size_t)MTOK * DI  * 2);  // dt -> ys
    unsigned short* YSUM  = (unsigned short*)alloc((size_t)MTOK * DI  * 2);

    auto cvt = [&](const float* s, unsigned short* d, int n) {
        cvt_f32_bf16<<<(n + 255) / 256, 256, 0, stream>>>(s, d, n);
    };

    // 1) weights -> bf16
    cvt(w_in,  WIN,  2 * DI * DM);
    cvt(w_out, WOUT, DM * DI);
    cvt(xp_f,  WXPF, XPN * DI);
    cvt(xp_b,  WXPB, XPN * DI);
    cvt(dtw_f, WDTF, DI * DTR);
    cvt(dtw_b, WDTB, DI * DTR);

    // 2) layernorm + residual copy
    layernorm_kernel<<<MTOK / 8, 256, 0, stream>>>(hs, nw, nb, HN, resid_p);

    // macro-tile: 32 rows x 64 cols per wave, 8 waves per block
    constexpr int TM = 2, TN = 4;
    auto gemm_blocks = [](int M, int N) { return (M / (16 * TM)) * (N / (16 * TN)) / 8; };

    // 3) in_proj: xz = HN[8192,512] * WIN[2048,512]^T -> bf16 [8192,2048]
    gemm_wmma_bf16<TM, TN, DM, DM, 2 * DI, true>
        <<<gemm_blocks(MTOK, 2 * DI), 256, 0, stream>>>(
            HN, WIN, XZ, (2 * DI) / (16 * TN), 1.0f);

    // 4) depthwise conv + SiLU (forward / reversed)
    {
        int blk = (MTOK * DI) / 256;
        conv_silu_kernel<<<blk, 256, 0, stream>>>(XZ, cw_f, cb_f, XCF, 0);
        conv_silu_kernel<<<blk, 256, 0, stream>>>(XZ, cw_b, cb_b, XCB, 1);
    }

    // 5) x_proj: x_dbl = XC[8192,1024] * WXP[64,1024]^T -> f32 [8192,64]
    gemm_wmma_bf16<TM, TN, DI, DI, XPN, false>
        <<<gemm_blocks(MTOK, XPN), 256, 0, stream>>>(
            XCF, WXPF, XDBLF, XPN / (16 * TN), 1.0f);
    gemm_wmma_bf16<TM, TN, DI, DI, XPN, false>
        <<<gemm_blocks(MTOK, XPN), 256, 0, stream>>>(
            XCB, WXPB, XDBLB, XPN / (16 * TN), 1.0f);

    // 6) pack dt_rank columns as bf16 A-matrix
    pack_dtin_kernel<<<(MTOK * DTR) / 256, 256, 0, stream>>>(XDBLF, DTINF);
    pack_dtin_kernel<<<(MTOK * DTR) / 256, 256, 0, stream>>>(XDBLB, DTINB);

    // 7) dt_proj: dt = DTIN[8192,32] * WDT[1024,32]^T -> bf16 [8192,1024]
    gemm_wmma_bf16<TM, TN, DTR, DTR, DI, true>
        <<<gemm_blocks(MTOK, DI), 256, 0, stream>>>(
            DTINF, WDTF, DTF, DI / (16 * TN), 1.0f);
    gemm_wmma_bf16<TM, TN, DTR, DTR, DI, true>
        <<<gemm_blocks(MTOK, DI), 256, 0, stream>>>(
            DTINB, WDTB, DTB, DI / (16 * TN), 1.0f);

    // 8) selective scans (dt buffer becomes y in place, logical order)
    scan_kernel<<<64, 256, 0, stream>>>(XCF, DTF, XDBLF, dtb_f, Al_f, Dk_f);
    scan_kernel<<<64, 256, 0, stream>>>(XCB, DTB, XDBLB, dtb_b, Al_b, Dk_b);

    // 9) combine: (y_f + rev(y_b)) * silu(z) -> bf16
    combine_kernel<<<(MTOK * DI) / 256, 256, 0, stream>>>(DTF, DTB, XZ, YSUM);

    // 10) out_proj with fused 0.5 scale: out = 0.5 * YSUM * WOUT^T -> f32
    gemm_wmma_bf16<TM, TN, DI, DI, DM, false>
        <<<gemm_blocks(MTOK, DM), 256, 0, stream>>>(
            YSUM, WOUT, out_p, DM / (16 * TN), 0.5f);

    (void)in_sizes; (void)n_in; (void)out_size; (void)ws_size;
}
